// GProjection3D_45217415692701
// MI455X (gfx1250) — compile-verified, matrix operand order checked by reading, not run on hardware
//
#include <hip/hip_runtime.h>

// GProjection3D for MI455X (gfx1250, wave32).
// Pipeline: prep(weights->f16 col-major) -> k1(center sample+shift matvec)
//        -> k2(27-nbr gather + fd1/fd2 MLP via v_wmma_f32_16x16x32_f16)
//        -> k3(agg GEMM K=1728 fused with center MLP via WMMA).

typedef __attribute__((ext_vector_type(16))) _Float16 v16h;
typedef __attribute__((ext_vector_type(8)))  float    v8f;

#define CC   64
#define DD   96
#define HH   96
#define WW   96
#define NB   27
#define K1P  96            // C+3=67 padded to 3*32
#define KAGG (NB * CC)     // 1728 = 54*32
#define DHW  (DD * HH * WW)

__device__ __forceinline__ v8f wmma16(v16h a, v16h b, v8f c) {
  // D = A(16x32 f16) * B(32x16 f16) + C(16x16 f32)
  return __builtin_amdgcn_wmma_f32_16x16x32_f16(false, a, false, b, (short)0, c,
                                                false, false);
}

// Load a 16-lane-indexed operand fragment (A rows or col-major B columns).
// Per ISA layout (16-bit 16x32): lane L(<16) row/col L, half-k group (L>>4)*8;
// elements are K = k0..k0+7 (vec 0..7) and K = k0+16..k0+23 (vec 8..15).
// Rows are K-contiguous and 16B aligned -> two 16-byte loads per lane.
__device__ __forceinline__ v16h load_frag(const _Float16* base, int stride,
                                          int rc0, int kbase, int lane) {
  const _Float16* p = base + (size_t)(rc0 + (lane & 15)) * stride + kbase +
                      ((lane >> 4) << 3);
  v16h f;
#pragma unroll
  for (int i = 0; i < 8; ++i) { f[i] = p[i]; f[i + 8] = p[i + 16]; }
  return f;
}

// Trilinear setup, padding_mode='border', align_corners=True.
__device__ __forceinline__ void tri_setup(float gx, float gy, float gz,
                                          int* off, float* wt) {
  float xs = fminf(fmaxf((gx + 1.f) * 0.5f * (WW - 1), 0.f), (float)(WW - 1));
  float ys = fminf(fmaxf((gy + 1.f) * 0.5f * (HH - 1), 0.f), (float)(HH - 1));
  float zs = fminf(fmaxf((gz + 1.f) * 0.5f * (DD - 1), 0.f), (float)(DD - 1));
  float xf = floorf(xs), yf = floorf(ys), zf = floorf(zs);
  float wx = xs - xf, wy = ys - yf, wz = zs - zf;
  int x0 = (int)xf, y0 = (int)yf, z0 = (int)zf;
  int x1 = min(x0 + 1, WW - 1), y1 = min(y0 + 1, HH - 1), z1 = min(z0 + 1, DD - 1);
  off[0] = (z0 * HH + y0) * WW + x0;  off[1] = (z0 * HH + y0) * WW + x1;
  off[2] = (z0 * HH + y1) * WW + x0;  off[3] = (z0 * HH + y1) * WW + x1;
  off[4] = (z1 * HH + y0) * WW + x0;  off[5] = (z1 * HH + y0) * WW + x1;
  off[6] = (z1 * HH + y1) * WW + x0;  off[7] = (z1 * HH + y1) * WW + x1;
  float ax = 1.f - wx, ay = 1.f - wy, az = 1.f - wz;
  wt[0] = az * ay * ax; wt[1] = az * ay * wx; wt[2] = az * wy * ax; wt[3] = az * wy * wx;
  wt[4] = wz * ay * ax; wt[5] = wz * ay * wx; wt[6] = wz * wy * ax; wt[7] = wz * wy * wx;
}

// ---- prep: f32 weights -> f16, K-padded, K-contiguous (col-major for B frags)
__global__ void k_prep_w(const float* __restrict__ src, _Float16* __restrict__ dst,
                         int O, int Kin, int Kpad) {
  int t = blockIdx.x * blockDim.x + threadIdx.x;
  if (t >= O * Kpad) return;
  int o = t / Kpad, k = t - o * Kpad;
  dst[t] = (_Float16)((k < Kin) ? src[o * Kin + k] : 0.f);
}

// sn_w [O][I][27] -> snw_cm [o][nb*64+i] f16 (K-contiguous)
__global__ void k_prep_snw(const float* __restrict__ src, _Float16* __restrict__ dst) {
  int t = blockIdx.x * blockDim.x + threadIdx.x;
  if (t >= CC * KAGG) return;
  int o = t / KAGG, rem = t - o * KAGG;
  int nb = rem >> 6, i = rem & 63;
  dst[t] = (_Float16)src[(o * CC + i) * NB + nb];
}

// ---- k1: center trilinear sample + shift_delta matvec -> 27 neighbor coords
__global__ void k_center_shift(const float* __restrict__ img,
                               const float* __restrict__ verts,
                               const float* __restrict__ sdw,
                               const float* __restrict__ sdb,
                               float* __restrict__ nbr, int n_verts) {
  int n = blockIdx.x * blockDim.x + threadIdx.x;
  if (n >= n_verts) return;
  float vx = fminf(fmaxf(verts[n * 3 + 0], -1.f), 1.f);
  float vy = fminf(fmaxf(verts[n * 3 + 1], -1.f), 1.f);
  float vz = fminf(fmaxf(verts[n * 3 + 2], -1.f), 1.f);
  int off[8]; float wt[8];
  tri_setup(vx, vy, vz, off, wt);
  float fc[CC];
#pragma unroll
  for (int c = 0; c < CC; ++c) {
    const float* p = img + (size_t)c * DHW;
    float a = 0.f;
#pragma unroll
    for (int k = 0; k < 8; ++k) a += wt[k] * p[off[k]];
    fc[c] = a;
  }
  float vc[3] = {vx, vy, vz};
  for (int j = 0; j < 81; ++j) {
    float a = sdb[j];
#pragma unroll
    for (int c = 0; c < CC; ++c) a += sdw[j * CC + c] * fc[c];
    int nb = j / 3, ax = j - nb * 3;
    float s = (nb == 0) ? 0.f : a;   // neighbor 0 keeps zero shift
    nbr[((size_t)n * NB + nb) * 3 + ax] = vc[ax] + s;
  }
}

// ---- k2: per-vertex 27-neighbor gather + fused fd1/fd2 MLP (WMMA)
__global__ void __launch_bounds__(128)
k_gather_mlp(const float* __restrict__ img, const float* __restrict__ nbr,
             const _Float16* __restrict__ w1, const float* __restrict__ b1,
             const _Float16* __restrict__ w2, const float* __restrict__ b2,
             _Float16* __restrict__ dbuf, _Float16* __restrict__ cenf) {
  int n = blockIdx.x;
  int tid = threadIdx.x;
  int lane = tid & 31, wave = tid >> 5;
  __shared__ int      s_off[NB][8];
  __shared__ float    s_wt[NB][8];
  __shared__ float    featf[NB][68];
  __shared__ _Float16 diffs[32][K1P];   // padded diff rows, f16
  __shared__ _Float16 h1s[32][CC];      // fd1 output (transposed store)

  if (tid < NB) {
    float gx = nbr[((size_t)n * NB + tid) * 3 + 0];
    float gy = nbr[((size_t)n * NB + tid) * 3 + 1];
    float gz = nbr[((size_t)n * NB + tid) * 3 + 2];
    tri_setup(gx, gy, gz, s_off[tid], s_wt[tid]);
    featf[tid][64] = gx; featf[tid][65] = gy; featf[tid][66] = gz;
  }
  __syncthreads();

  // Gather: 27 neighbors x 64 channels, trilinear blend.
  for (int t = tid; t < NB * CC; t += 128) {
    int nb = t >> 6, c = t & 63;
    const float* p = img + (size_t)c * DHW;
    float a = 0.f;
#pragma unroll
    for (int k = 0; k < 8; ++k) a += s_wt[nb][k] * p[s_off[nb][k]];
    featf[nb][c] = a;
  }
  __syncthreads();

  // diff rows (f16, zero-padded to 32x96) and center-feature row (nb==13).
  for (int t = tid; t < 32 * K1P; t += 128) {
    int r = t / K1P, k = t - r * K1P;
    float v = (r < NB && k < 67) ? (featf[r][k] - featf[0][k]) : 0.f;
    diffs[r][k] = (_Float16)v;
  }
  if (tid < K1P) {
    float v = (tid < 67) ? featf[13][tid] : 0.f;
    cenf[(size_t)n * K1P + tid] = (_Float16)v;
  }
  __syncthreads();

  int n0 = wave * 16;                       // each wave owns one 16-col tile
  float bb1 = b1[n0 + (lane & 15)];
#pragma unroll
  for (int mt = 0; mt < 2; ++mt) {          // fd1: rows 0..31, K=96
    v8f acc = {};
#pragma unroll
    for (int ks = 0; ks < 3; ++ks) {
      v16h a = load_frag(&diffs[0][0], K1P, mt * 16, ks * 32, lane);
      v16h b = load_frag(w1, K1P, n0, ks * 32, lane);
      acc = wmma16(a, b, acc);
    }
#pragma unroll
    for (int r = 0; r < 8; ++r) {           // transpose via LDS (D layout)
      int M = mt * 16 + r + ((lane >> 4) << 3);
      h1s[M][n0 + (lane & 15)] = (_Float16)(acc[r] + bb1);
    }
  }
  __syncthreads();

  float bb2 = b2[n0 + (lane & 15)];
#pragma unroll
  for (int mt = 0; mt < 2; ++mt) {          // fd2: K=64
    v8f acc = {};
#pragma unroll
    for (int ks = 0; ks < 2; ++ks) {
      v16h a = load_frag(&h1s[0][0], CC, mt * 16, ks * 32, lane);
      v16h b = load_frag(w2, CC, n0, ks * 32, lane);
      acc = wmma16(a, b, acc);
    }
#pragma unroll
    for (int r = 0; r < 8; ++r) {
      int M = mt * 16 + r + ((lane >> 4) << 3);
      if (M < NB)                           // d row layout: [n][nb*64 + c]
        dbuf[(size_t)n * KAGG + M * CC + n0 + (lane & 15)] =
            (_Float16)(acc[r] + bb2);
    }
  }
}

// ---- k3: fused center MLP + agg GEMM (K=1728) -> out [N][64] f32
__global__ void __launch_bounds__(128)
k_out(const _Float16* __restrict__ dbuf, const _Float16* __restrict__ snw,
      const float* __restrict__ snb, const _Float16* __restrict__ cenf,
      const _Float16* __restrict__ w1c, const float* __restrict__ b1c,
      const _Float16* __restrict__ w2c, const float* __restrict__ b2c,
      float* __restrict__ out) {
  int v0 = blockIdx.x * 16;
  int tid = threadIdx.x, lane = tid & 31, wave = tid >> 5;
  int n0 = wave * 16;
  __shared__ _Float16 hc[16][CC];

  // center layer 1: [16 x 96] x [96 x 64]
  v8f h = {};
  const _Float16* abase = cenf + (size_t)v0 * K1P;
#pragma unroll
  for (int ks = 0; ks < 3; ++ks) {
    v16h a = load_frag(abase, K1P, 0, ks * 32, lane);
    v16h b = load_frag(w1c, K1P, n0, ks * 32, lane);
    h = wmma16(a, b, h);
  }
  float bb1 = b1c[n0 + (lane & 15)];
#pragma unroll
  for (int r = 0; r < 8; ++r) {
    int M = r + ((lane >> 4) << 3);
    hc[M][n0 + (lane & 15)] = (_Float16)(h[r] + bb1);
  }
  __syncthreads();

  v8f acc = {};
#pragma unroll
  for (int ks = 0; ks < 2; ++ks) {          // center layer 2: K=64
    v16h a = load_frag(&hc[0][0], CC, 0, ks * 32, lane);
    v16h b = load_frag(w2c, CC, n0, ks * 32, lane);
    acc = wmma16(a, b, acc);
  }
  // agg GEMM: [16 x 1728] x [1728 x 64]; A streamed, B L2-resident.
  const _Float16* dbase = dbuf + (size_t)v0 * KAGG;
  for (int ks = 0; ks < KAGG / 32; ++ks) {
    if (ks + 2 < KAGG / 32)                 // global_prefetch_b8 on A stream
      __builtin_prefetch(dbase + (size_t)(lane & 15) * KAGG + (ks + 2) * 32, 0, 0);
    v16h a = load_frag(dbase, KAGG, 0, ks * 32, lane);
    v16h b = load_frag(snw, KAGG, n0, ks * 32, lane);
    acc = wmma16(a, b, acc);
  }
  float bb = b2c[n0 + (lane & 15)] + snb[n0 + (lane & 15)];
#pragma unroll
  for (int r = 0; r < 8; ++r) {
    int M = r + ((lane >> 4) << 3);
    out[(size_t)(v0 + M) * CC + n0 + (lane & 15)] = acc[r] + bb;
  }
}

extern "C" void kernel_launch(void* const* d_in, const int* in_sizes, int n_in,
                              void* d_out, int out_size, void* d_ws, size_t ws_size,
                              hipStream_t stream) {
  (void)n_in; (void)out_size; (void)ws_size;
  const float* images = (const float*)d_in[0];
  const float* verts  = (const float*)d_in[1];
  const float* sd_w   = (const float*)d_in[2];
  const float* sd_b   = (const float*)d_in[3];
  const float* fd1_w  = (const float*)d_in[4];
  const float* fd1_b  = (const float*)d_in[5];
  const float* fd2_w  = (const float*)d_in[6];
  const float* fd2_b  = (const float*)d_in[7];
  const float* fc1_w  = (const float*)d_in[8];
  const float* fc1_b  = (const float*)d_in[9];
  const float* fc2_w  = (const float*)d_in[10];
  const float* fc2_b  = (const float*)d_in[11];
  const float* sn_w   = (const float*)d_in[12];
  const float* sn_b   = (const float*)d_in[13];
  const int n_verts = in_sizes[1] / 3;

  char* ws = (char*)d_ws;
  size_t off = 0;
  auto take = [&](size_t bytes) {
    void* p = ws + off;
    off = (off + bytes + 255) & ~(size_t)255;
    return p;
  };
  float*    nbr  = (float*)take((size_t)n_verts * NB * 3 * sizeof(float));
  _Float16* w1d  = (_Float16*)take((size_t)CC * K1P * 2);
  _Float16* w2d  = (_Float16*)take((size_t)CC * CC * 2);
  _Float16* w1c  = (_Float16*)take((size_t)CC * K1P * 2);
  _Float16* w2c  = (_Float16*)take((size_t)CC * CC * 2);
  _Float16* snw  = (_Float16*)take((size_t)CC * KAGG * 2);
  _Float16* cenf = (_Float16*)take((size_t)n_verts * K1P * 2);
  _Float16* dbuf = (_Float16*)take((size_t)n_verts * KAGG * 2);

  k_prep_w<<<(CC * K1P + 255) / 256, 256, 0, stream>>>(fd1_w, w1d, CC, CC + 3, K1P);
  k_prep_w<<<(CC * CC  + 255) / 256, 256, 0, stream>>>(fd2_w, w2d, CC, CC, CC);
  k_prep_w<<<(CC * K1P + 255) / 256, 256, 0, stream>>>(fc1_w, w1c, CC, CC + 3, K1P);
  k_prep_w<<<(CC * CC  + 255) / 256, 256, 0, stream>>>(fc2_w, w2c, CC, CC, CC);
  k_prep_snw<<<(CC * KAGG + 255) / 256, 256, 0, stream>>>(sn_w, snw);

  k_center_shift<<<(n_verts + 255) / 256, 256, 0, stream>>>(
      images, verts, sd_w, sd_b, nbr, n_verts);

  k_gather_mlp<<<n_verts, 128, 0, stream>>>(
      images, nbr, w1d, fd1_b, w2d, fd2_b, dbuf, cenf);

  k_out<<<n_verts / 16, 128, 0, stream>>>(
      dbuf, snw, sn_b, cenf, w1c, fc1_b, w2c, fc2_b, (float*)d_out);
}
